// GATModel_15118284882415
// MI455X (gfx1250) — compile-verified, matrix operand order checked by reading, not run on hardware
//
#include <hip/hip_runtime.h>
#include <hip/hip_bf16.h>

typedef __attribute__((ext_vector_type(16))) _Float16 v16h;
typedef __attribute__((ext_vector_type(8)))  float    v8f;
typedef __attribute__((ext_vector_type(4)))  int      v4i;

#define NEG_SLOPE 0.2f

#define GLOBAL_AS __attribute__((address_space(1)))
#define LDS_AS    __attribute__((address_space(3)))

#if defined(__has_builtin)
#if __has_builtin(__builtin_amdgcn_global_load_async_to_lds_b128) && \
    __has_builtin(__builtin_amdgcn_s_wait_asynccnt)
#define USE_ASYNC_LDS 1
#endif
#endif

// Copy 8 f16 (16B) from global to LDS: async DMA path on gfx1250, vector fallback.
__device__ __forceinline__ void copy16B_to_lds(const _Float16* g, _Float16* l)
{
#ifdef USE_ASYNC_LDS
    __builtin_amdgcn_global_load_async_to_lds_b128(
        (GLOBAL_AS v4i*)g, (LDS_AS v4i*)l, 0, 0);
#else
    *(uint4*)l = *(const uint4*)g;
#endif
}

__device__ __forceinline__ void wait_lds_copies()
{
#ifdef USE_ASYNC_LDS
    __builtin_amdgcn_s_wait_asynccnt(0);
#endif
}

// ---------------------------------------------------------------------------
// WMMA GEMM: C[M,Ncols] = A[M,K] * Bt[Ncols,K]^T, f16 inputs, f32 accumulate.
// Both operands are K-major, so tiles are [rows][32] 16B-aligned slabs.
// ---------------------------------------------------------------------------
template <int BM, int NWAVES>
__global__ __launch_bounds__(NWAVES * 32) void gemm_wmma_f16(
    const _Float16* __restrict__ A, const _Float16* __restrict__ Bt,
    float* __restrict__ Cm, int M, int K, int Ncols)
{
    constexpr int BN  = NWAVES * 16;
    constexpr int LDA = 40;                       // 32 K-elems + pad (80B rows, 16B aligned)
    __shared__ alignas(16) _Float16 As[BM * LDA];
    __shared__ alignas(16) _Float16 Bs[BN * LDA];

    const int tid  = threadIdx.x;
    const int nthr = NWAVES * 32;
    const int wave = tid >> 5;
    const int lane = tid & 31;
    const int m0 = blockIdx.x * BM;
    const int n0 = blockIdx.y * BN;
    const int lr = lane & 15;            // row (A) / col (B) within 16-tile
    const int kb = (lane >> 4) * 8;      // K base for this half-wave

    v8f acc[BM / 16] = {};

    for (int k0 = 0; k0 < K; k0 += 32) {
        // A tile: BM rows x 32 K, 16B per element-copy (no divergence: clamp row)
        for (int i = tid; i < BM * 4; i += nthr) {
            int r = i >> 2, c8 = (i & 3) * 8;
            int gr = m0 + r; gr = gr < M ? gr : M - 1;
            copy16B_to_lds(A + (long)gr * K + k0 + c8, &As[r * LDA + c8]);
        }
        // B tile: BN rows of Bt (Ncols-major), same slab shape
        for (int i = tid; i < BN * 4; i += nthr) {
            int r = i >> 2, c8 = (i & 3) * 8;
            int gn = n0 + r; gn = gn < Ncols ? gn : Ncols - 1;
            copy16B_to_lds(Bt + (long)gn * K + k0 + c8, &Bs[r * LDA + c8]);
        }
        wait_lds_copies();
        __syncthreads();

        // Fragments (ISA 7.12.2): elem i -> K = kb + i (+8 if i>=8)
        v16h bf;
        {
            const _Float16* bp = &Bs[(wave * 16 + lr) * LDA + kb];
#pragma unroll
            for (int i = 0; i < 16; ++i) bf[i] = bp[i + ((i >= 8) ? 8 : 0)];
        }
#pragma unroll
        for (int s = 0; s < BM / 16; ++s) {
            v16h af;
            const _Float16* ap = &As[(s * 16 + lr) * LDA + kb];
#pragma unroll
            for (int i = 0; i < 16; ++i) af[i] = ap[i + ((i >= 8) ? 8 : 0)];
            acc[s] = __builtin_amdgcn_wmma_f32_16x16x32_f16(
                false, af, false, bf, (short)0, acc[s], false, false);
        }
        __syncthreads();
    }

    // C/D layout: VGPR r -> row r + 8*(lane>=16), col = lane&15
    const int gn = n0 + wave * 16 + lr;
#pragma unroll
    for (int s = 0; s < BM / 16; ++s) {
        int mb = m0 + s * 16 + ((lane >> 4) ? 8 : 0);
#pragma unroll
        for (int r = 0; r < 8; ++r) {
            int gm = mb + r;
            if (gm < M && gn < Ncols) Cm[(long)gm * Ncols + gn] = acc[s][r];
        }
    }
}

// ---------------------------------------------------------------------------
// Conversion / utility kernels
// ---------------------------------------------------------------------------
__global__ void cvt_f16_k(const float* __restrict__ s, _Float16* __restrict__ d, long n)
{
    long i = blockIdx.x * (long)blockDim.x + threadIdx.x;
    if (i < n) d[i] = (_Float16)s[i];
}

// W[K,Ncols] fp32 -> Wt[Ncols,K] f16
__global__ void transpose_cvt_k(const float* __restrict__ W, _Float16* __restrict__ Wt,
                                int K, int Ncols)
{
    int idx = blockIdx.x * blockDim.x + threadIdx.x;
    if (idx >= K * Ncols) return;
    int k = idx / Ncols, n = idx % Ncols;
    Wt[(long)n * K + k] = (_Float16)W[idx];
}

__global__ void fill_zero_u32(unsigned* __restrict__ p, long n)
{
    long i = blockIdx.x * (long)blockDim.x + threadIdx.x;
    if (i < n) p[i] = 0u;
}

// a_s[n,h] = sum_c h[n,h,c]*att_s[h,c]; same for a_d
__global__ void attn_scores_k(const float* __restrict__ h,
                              const float* __restrict__ atts,
                              const float* __restrict__ attd,
                              float* __restrict__ as, float* __restrict__ ad,
                              int N, int H, int C)
{
    int idx = blockIdx.x * blockDim.x + threadIdx.x;
    if (idx >= N * H) return;
    int n = idx / H, hh = idx % H;
    const float* hp = h + (long)n * H * C + hh * C;
    float ss = 0.0f, sd = 0.0f;
    for (int c = 0; c < C; ++c) {
        float v = hp[c];
        ss += v * atts[hh * C + c];
        sd += v * attd[hh * C + c];
    }
    as[idx] = ss;
    ad[idx] = sd;
}

// monotonic float<->uint encoding for atomicMax on possibly-negative scores
__device__ __forceinline__ unsigned fenc(float x)
{
    unsigned u = __float_as_uint(x);
    return (u & 0x80000000u) ? ~u : (u | 0x80000000u);
}
__device__ __forceinline__ float fdec(unsigned u)
{
    return __uint_as_float((u & 0x80000000u) ? (u & 0x7FFFFFFFu) : ~u);
}

__global__ void edge_max_k(const int* __restrict__ ei, int E, int N,
                           const float* __restrict__ as, const float* __restrict__ ad,
                           unsigned* __restrict__ mu, int H)
{
    long idx = blockIdx.x * (long)blockDim.x + threadIdx.x;
    long ET  = (long)E + N;
    if (idx >= ET * H) return;
    int e  = (int)(idx / H);
    int hh = (int)(idx % H);
    int s = (e < E) ? ei[e] : (e - E);
    int d = (e < E) ? ei[E + e] : (e - E);
    float sc = as[s * H + hh] + ad[d * H + hh];
    sc = sc > 0.0f ? sc : NEG_SLOPE * sc;
    atomicMax(&mu[d * H + hh], fenc(sc));
}

__global__ void decode_k(float* __restrict__ p, long n)
{
    long i = blockIdx.x * (long)blockDim.x + threadIdx.x;
    if (i < n) p[i] = fdec(__float_as_uint(p[i]));
}

__global__ void edge_expsum_k(const int* __restrict__ ei, int E, int N,
                              const float* __restrict__ as, const float* __restrict__ ad,
                              const float* __restrict__ m, float* __restrict__ dn, int H)
{
    long idx = blockIdx.x * (long)blockDim.x + threadIdx.x;
    long ET  = (long)E + N;
    if (idx >= ET * H) return;
    int e  = (int)(idx / H);
    int hh = (int)(idx % H);
    int s = (e < E) ? ei[e] : (e - E);
    int d = (e < E) ? ei[E + e] : (e - E);
    float sc = as[s * H + hh] + ad[d * H + hh];
    sc = sc > 0.0f ? sc : NEG_SLOPE * sc;
    atomicAdd(&dn[d * H + hh], __expf(sc - m[d * H + hh]));
}

// One wave32 per edge; lanes stride the H*C channels. Source rows and the
// accumulator are L2-resident (51MB/12.8MB << 192MB global L2).
__global__ void edge_agg_k(const int* __restrict__ ei, int E, int N,
                           const float* __restrict__ h,
                           const float* __restrict__ as, const float* __restrict__ ad,
                           const float* __restrict__ m, const float* __restrict__ dn,
                           float* __restrict__ out, int H, int C)
{
    long wid = (blockIdx.x * (long)blockDim.x + threadIdx.x) >> 5;
    int lane = threadIdx.x & 31;
    long ET = (long)E + N;
    if (wid >= ET) return;
    int e = (int)wid;
    int s = (e < E) ? ei[e] : (e - E);
    int d = (e < E) ? ei[E + e] : (e - E);
    int Cf = H * C;
    const float* hs = h + (long)s * Cf;
    float* od = out + (long)d * Cf;
    __builtin_prefetch(hs, 0, 0);   // global_prefetch_b8 on the random-gather row
    for (int j = lane; j < Cf; j += 32) {
        int hh = j / C;
        float sc = as[s * H + hh] + ad[d * H + hh];
        sc = sc > 0.0f ? sc : NEG_SLOPE * sc;
        float alpha = __expf(sc - m[d * H + hh]) / (dn[d * H + hh] + 1e-16f);
        atomicAdd(&od[j], alpha * hs[j]);
    }
}

// act1h[n,j] = f16(elu(agg[n,j] + bias[j]))   (layer-1 epilogue feeding GEMM2)
__global__ void bias_elu_f16_k(const float* __restrict__ a, const float* __restrict__ bias,
                               _Float16* __restrict__ d, long total, int Cf)
{
    long i = blockIdx.x * (long)blockDim.x + threadIdx.x;
    if (i >= total) return;
    int j = (int)(i % Cf);
    float v = a[i] + bias[j];
    v = v > 0.0f ? v : (__expf(v) - 1.0f);
    d[i] = (_Float16)v;
}

// out[n,j] += bias[j]   (final epilogue)
__global__ void bias_add_k(float* __restrict__ a, const float* __restrict__ bias,
                           long total, int Cf)
{
    long i = blockIdx.x * (long)blockDim.x + threadIdx.x;
    if (i >= total) return;
    a[i] += bias[(int)(i % Cf)];
}

// ---------------------------------------------------------------------------
// Launcher
// ---------------------------------------------------------------------------
static inline unsigned nblk(long n, int b) { return (unsigned)((n + b - 1) / b); }

extern "C" void kernel_launch(void* const* d_in, const int* in_sizes, int n_in,
                              void* d_out, int out_size, void* d_ws, size_t ws_size,
                              hipStream_t stream)
{
    const float* x     = (const float*)d_in[0];
    const int*   ei    = (const int*)d_in[1];
    const float* W1    = (const float*)d_in[2];
    const float* atts1 = (const float*)d_in[3];
    const float* attd1 = (const float*)d_in[4];
    const float* b1    = (const float*)d_in[5];
    const float* W2    = (const float*)d_in[6];
    const float* atts2 = (const float*)d_in[7];
    const float* attd2 = (const float*)d_in[8];
    const float* b2    = (const float*)d_in[9];
    float* out = (float*)d_out;

    const int Din = 128, H1 = 4, C1 = 64, Cf1 = H1 * C1, C2 = 64;
    const int N = in_sizes[0] / Din;
    const int E = in_sizes[1] / 2;
    const long ET = (long)E + N;

    // workspace layout
    float* ws   = (float*)d_ws;
    float* h1   = ws;                   // N*256 f32
    float* agg1 = h1 + (long)N * Cf1;   // N*256 f32
    float* h2   = agg1 + (long)N * Cf1; // N*64  f32
    float* as1  = h2 + (long)N * C2;    // N*4
    float* ad1  = as1 + (long)N * H1;
    float* m1   = ad1 + (long)N * H1;
    float* dn1  = m1 + (long)N * H1;
    float* as2  = dn1 + (long)N * H1;   // N
    float* ad2  = as2 + N;
    float* m2   = ad2 + N;
    float* dn2  = m2 + N;
    _Float16* xh    = (_Float16*)(dn2 + N);        // N*128 f16
    _Float16* act1h = xh + (long)N * Din;          // N*256 f16
    _Float16* w1t   = act1h + (long)N * Cf1;       // 256*128 f16 (transposed)
    _Float16* w2t   = w1t + (long)Cf1 * Din;       // 64*256 f16 (transposed)

    // ---------------- Layer 1 ----------------
    cvt_f16_k<<<nblk((long)N * Din, 256), 256, 0, stream>>>(x, xh, (long)N * Din);
    transpose_cvt_k<<<nblk((long)Din * Cf1, 256), 256, 0, stream>>>(W1, w1t, Din, Cf1);

    // h1 = x @ W1   (50000x128 @ 128x256)
    gemm_wmma_f16<64, 8><<<dim3(nblk(N, 64), 2), 256, 0, stream>>>(xh, w1t, h1, N, Din, Cf1);

    attn_scores_k<<<nblk((long)N * H1, 256), 256, 0, stream>>>(h1, atts1, attd1, as1, ad1, N, H1, C1);

    fill_zero_u32<<<nblk((long)N * H1 * 2, 256), 256, 0, stream>>>((unsigned*)m1, (long)N * H1 * 2); // m1+dn1
    fill_zero_u32<<<nblk((long)N * Cf1, 256), 256, 0, stream>>>((unsigned*)agg1, (long)N * Cf1);

    edge_max_k<<<nblk(ET * H1, 256), 256, 0, stream>>>(ei, E, N, as1, ad1, (unsigned*)m1, H1);
    decode_k<<<nblk((long)N * H1, 256), 256, 0, stream>>>(m1, (long)N * H1);
    edge_expsum_k<<<nblk(ET * H1, 256), 256, 0, stream>>>(ei, E, N, as1, ad1, m1, dn1, H1);
    edge_agg_k<<<nblk(ET * 32, 256), 256, 0, stream>>>(ei, E, N, h1, as1, ad1, m1, dn1, agg1, H1, C1);

    // act1h = f16(elu(agg1 + bias1))
    bias_elu_f16_k<<<nblk((long)N * Cf1, 256), 256, 0, stream>>>(agg1, b1, act1h, (long)N * Cf1, Cf1);

    // ---------------- Layer 2 ----------------
    transpose_cvt_k<<<nblk((long)Cf1 * C2, 256), 256, 0, stream>>>(W2, w2t, Cf1, C2);

    // h2 = act1 @ W2  (50000x256 @ 256x64)
    gemm_wmma_f16<64, 4><<<dim3(nblk(N, 64), 1), 128, 0, stream>>>(act1h, w2t, h2, N, Cf1, C2);

    attn_scores_k<<<nblk(N, 256), 256, 0, stream>>>(h2, atts2, attd2, as2, ad2, N, 1, C2);

    fill_zero_u32<<<nblk((long)N * 2, 256), 256, 0, stream>>>((unsigned*)m2, (long)N * 2);   // m2+dn2
    fill_zero_u32<<<nblk((long)N * C2, 256), 256, 0, stream>>>((unsigned*)out, (long)N * C2);

    edge_max_k<<<nblk(ET, 256), 256, 0, stream>>>(ei, E, N, as2, ad2, (unsigned*)m2, 1);
    decode_k<<<nblk(N, 256), 256, 0, stream>>>(m2, N);
    edge_expsum_k<<<nblk(ET, 256), 256, 0, stream>>>(ei, E, N, as2, ad2, m2, dn2, 1);
    edge_agg_k<<<nblk(ET * 32, 256), 256, 0, stream>>>(ei, E, N, h2, as2, ad2, m2, dn2, out, 1, C2);

    bias_add_k<<<nblk((long)N * C2, 256), 256, 0, stream>>>(out, b2, (long)N * C2, C2);
}
